// ContextQueryAttention_58351425683783
// MI455X (gfx1250) — matrix operand order repaired and checked by reading, not compiled, not run on hardware
//
#include <hip/hip_runtime.h>
#include <math.h>

// ---------------------------------------------------------------------------
// Context-Query attention for MI455X (gfx1250, wave32, WMMA bf16 16x16x32,
// async global->LDS double-buffered B-panel staging).
// ---------------------------------------------------------------------------

typedef __bf16 bf16;
typedef __attribute__((ext_vector_type(4)))  bf16  v4bf;
typedef __attribute__((ext_vector_type(8)))  bf16  v8bf;
typedef __attribute__((ext_vector_type(16))) bf16  v16bf;
typedef __attribute__((ext_vector_type(8)))  float v8f;

static constexpr int BATCH = 16;
static constexpr int LEN_C = 2048;  // L1 (context)
static constexpr int LEN_Q = 512;   // L2 (query)
static constexpr int DIM   = 128;   // D

// B-panel in LDS: 64 rows x 32 k (bf16), rows padded 64B -> 80B (bank spread,
// keeps 16B alignment). Two buffers for double buffering.
static constexpr int BROW_B  = 80;              // bytes per padded row
static constexpr int BROW_E  = BROW_B / 2;      // bf16 elems per padded row
static constexpr int BPANEL_E = 64 * BROW_E;    // elems per buffer (2560)
static constexpr int BPANEL_B = 64 * BROW_B;    // bytes per buffer (5120)

// ---- WMMA fragment helpers (contiguous-k lane layouts, wave32) --------------

// A-matrix 16x32 bf16: lanes 0-15 hold K=0..15;
//                      lanes 16-31 hold K=16..31.  m = lane&15.
__device__ __forceinline__ v16bf load_matA(const bf16* __restrict__ base, int ld,
                                           int row, int kk, int h) {
  const bf16* p = base + (size_t)row * ld + kk + 16 * h;
  v8bf lo = *(const v8bf*)(p);
  v8bf hi = *(const v8bf*)(p + 8);
  return __builtin_shufflevector(lo, hi, 0, 1, 2, 3, 4, 5, 6, 7,
                                 8, 9, 10, 11, 12, 13, 14, 15);
}

// B-matrix 32x16 bf16: lanes 0-15 hold K=0..15, lanes 16-31 hold K=16..31.
// Fragment = 16 contiguous bf16 from the padded LDS row.
__device__ __forceinline__ v16bf load_matB_lds(const bf16* p) {
  v8bf lo = *(const v8bf*)(p);
  v8bf hi = *(const v8bf*)(p + 8);
  return __builtin_shufflevector(lo, hi, 0, 1, 2, 3, 4, 5, 6, 7,
                                 8, 9, 10, 11, 12, 13, 14, 15);
}

__device__ __forceinline__ v8f wmma_bf16(v16bf a, v16bf b, v8f c) {
  return __builtin_amdgcn_wmma_f32_16x16x32_bf16(false, a, false, b,
                                                 (short)0, c, false, false);
}

// Async copy of one 16-byte chunk into LDS (per-lane; tracked by ASYNCcnt).
__device__ __forceinline__ void async_b128_to_lds(unsigned lds_byte_addr,
                                                  const bf16* g) {
  asm volatile("global_load_async_to_lds_b128 %0, %1, off"
               :: "v"(lds_byte_addr),
                  "v"((unsigned long long)(uintptr_t)g)
               : "memory");
}

// ---------------------------------------------------------------------------
// Block-cooperative GEMM: 8 waves compute a 128x64 tile of A(MxK) x B^T where
// the B source is stored row-major (N x K). The 64-row B panel (4KB/k-step)
// is staged in LDS by one async instruction per wave and double-buffered:
// panel k+1 streams in while panel k feeds the WMMAs. All four B fragments
// are materialized in registers first so the ds_loads issue as one clause
// and drain with graduated DScnt waits under the WMMA chain.
// ---------------------------------------------------------------------------
template <int KSTEPS>
__device__ __forceinline__ void gemm_lds_tile(
    const bf16* __restrict__ A, int lda, int m0,
    const bf16* __restrict__ Bsrc, int ldb, int n0,
    bf16* Btile, int tid, v8f acc[4]) {
  const int lane = tid & 31;
  const int h = lane >> 4, r = lane & 15;
  const int row   = tid >> 2;  // 0..63: B row this thread stages
  const int chunk = tid & 3;   // which 16B chunk of the 64B row
  const bf16* gsrc = Bsrc + (size_t)(n0 + row) * ldb + chunk * 8;
  // Generic LDS pointer -> byte offset inside LDS aperture (addr[31:0]).
  const unsigned lds0 =
      (unsigned)(uintptr_t)Btile + (unsigned)(row * BROW_B + chunk * 16);

  async_b128_to_lds(lds0, gsrc);  // prologue: panel 0
  for (int ks = 0; ks < KSTEPS; ++ks) {
    const int cur = ks & 1;
    if (ks + 1 < KSTEPS) {
      async_b128_to_lds(lds0 + (cur ^ 1) * BPANEL_B, gsrc + (ks + 1) * 32);
      asm volatile("s_wait_asynccnt 0x1" ::: "memory");  // current panel done
    } else {
      asm volatile("s_wait_asynccnt 0x0" ::: "memory");
    }
    __syncthreads();  // panel `cur` visible to all waves

    v16bf av = load_matA(A, lda, m0 + r, ks * 32, h);
    const bf16* bt = Btile + cur * BPANEL_E;
    v16bf bv[4];
#pragma unroll
    for (int nt = 0; nt < 4; ++nt)
      bv[nt] = load_matB_lds(bt + (nt * 16 + r) * BROW_E + 16 * h);
#pragma unroll
    for (int nt = 0; nt < 4; ++nt)
      acc[nt] = wmma_bf16(av, bv[nt], acc[nt]);
    __syncthreads();  // all reads done before panel `cur` is overwritten
  }
}

// ---------------------------------------------------------------------------
// Kernel 1: biases cw = C@wc, qw = Q@wq; bf16 operand copies.
//   Abf (B,L1,D) = bf16(ctx * wcq)   -> A of score GEMM
//   Qbf (B,L2,D) = bf16(query)       -> B of score GEMM (row-major NxK)
//   QT  (B,D,L2) = bf16(query)^T     -> B of a-GEMM
//   CT  (B,D,L1) = bf16(ctx)^T       -> B of t-GEMM
// ---------------------------------------------------------------------------
__global__ __launch_bounds__(256) void cqa_prep(
    const float* __restrict__ ctx, const float* __restrict__ qry,
    const float* __restrict__ w,
    bf16* __restrict__ Abf, bf16* __restrict__ Qbf,
    bf16* __restrict__ QT, bf16* __restrict__ CT,
    float* __restrict__ cw, float* __restrict__ qw) {
  int wid  = (blockIdx.x * blockDim.x + threadIdx.x) >> 5;
  int lane = threadIdx.x & 31;
  int d0   = lane * 4;
  const int rows_c     = BATCH * LEN_C;
  const int rows_total = rows_c + BATCH * LEN_Q;
  if (wid >= rows_total) return;

  if (wid < rows_c) {  // context rows
    int b = wid / LEN_C, i = wid - b * LEN_C;
    float4 v = *(const float4*)(ctx + (size_t)wid * DIM + d0);
    const float* wc  = w + DIM;
    const float* wcq = w + 2 * DIM;
    float dot = v.x * wc[d0] + v.y * wc[d0 + 1] + v.z * wc[d0 + 2] + v.w * wc[d0 + 3];
#pragma unroll
    for (int off = 16; off > 0; off >>= 1) dot += __shfl_xor(dot, off, 32);
    if (lane == 0) cw[wid] = dot;
    v4bf a;
    a[0] = (bf16)(v.x * wcq[d0]);     a[1] = (bf16)(v.y * wcq[d0 + 1]);
    a[2] = (bf16)(v.z * wcq[d0 + 2]); a[3] = (bf16)(v.w * wcq[d0 + 3]);
    *(v4bf*)(Abf + (size_t)wid * DIM + d0) = a;
    float comp[4] = {v.x, v.y, v.z, v.w};
#pragma unroll
    for (int t = 0; t < 4; ++t)
      CT[((size_t)b * DIM + d0 + t) * LEN_C + i] = (bf16)comp[t];
  } else {  // query rows
    int qr = wid - rows_c;
    int b = qr / LEN_Q, j = qr - b * LEN_Q;
    float4 v = *(const float4*)(qry + (size_t)qr * DIM + d0);
    const float* wqv = w;
    float dot = v.x * wqv[d0] + v.y * wqv[d0 + 1] + v.z * wqv[d0 + 2] + v.w * wqv[d0 + 3];
#pragma unroll
    for (int off = 16; off > 0; off >>= 1) dot += __shfl_xor(dot, off, 32);
    if (lane == 0) qw[qr] = dot;
    v4bf q;
    q[0] = (bf16)v.x; q[1] = (bf16)v.y; q[2] = (bf16)v.z; q[3] = (bf16)v.w;
    *(v4bf*)(Qbf + (size_t)qr * DIM + d0) = q;
    float comp[4] = {v.x, v.y, v.z, v.w};
#pragma unroll
    for (int t = 0; t < 4; ++t)
      QT[((size_t)b * DIM + d0 + t) * LEN_Q + j] = (bf16)comp[t];
  }
}

// ---------------------------------------------------------------------------
// Kernel 2: S[b,i,j] = (ctx*wcq)[i].q[j] + cw[i] + qw[j]  (f32, 64 MiB in L2)
// Block = 128x64 tile, K = D = 128.
// ---------------------------------------------------------------------------
__global__ __launch_bounds__(256) void cqa_scores(
    const bf16* __restrict__ Abf, const bf16* __restrict__ Qbf,
    const float* __restrict__ cw, const float* __restrict__ qw,
    float* __restrict__ S) {
  __shared__ __align__(16) bf16 Btile[2 * BPANEL_E];
  int tid = threadIdx.x;
  int lane = tid & 31, wave = tid >> 5;
  int h = lane >> 4, r = lane & 15;
  int bi = blockIdx.x;
  int b  = bi >> 7;            // 16 mb * 8 jb = 128 blocks per batch
  int mb = (bi >> 3) & 15;
  int jb = bi & 7;
  int i0 = mb * 128 + wave * 16;
  int j0 = jb * 64;
  const bf16* A = Abf + (size_t)b * LEN_C * DIM;
  const bf16* Q = Qbf + (size_t)b * LEN_Q * DIM;
  v8f acc[4] = {};
  gemm_lds_tile<DIM / 32>(A, DIM, i0, Q, DIM, j0, Btile, tid, acc);
  const float* cwb = cw + b * LEN_C;
  const float* qwb = qw + b * LEN_Q;
  float* Sb = S + (size_t)b * LEN_C * LEN_Q;
#pragma unroll
  for (int nt = 0; nt < 4; ++nt) {
    int n = j0 + nt * 16 + r;
    float qn = qwb[n];
#pragma unroll
    for (int rr = 0; rr < 8; ++rr) {
      int m = i0 + rr + 8 * h;
      Sb[(size_t)m * LEN_Q + n] = acc[nt][rr] + cwb[m] + qn;
    }
  }
}

// ---------------------------------------------------------------------------
// Kernel 3: column softmax over i (axis=1) for s1 (query mask) and s2
// (context mask). Block = 64 columns x 4 i-strips. Emits bf16 s1 (row-major)
// and bf16 s2^T (so the t-GEMM A operand is row-major).
// ---------------------------------------------------------------------------
__global__ __launch_bounds__(256) void cqa_softmax(
    const float* __restrict__ S, const int* __restrict__ mask_p,
    const int* __restrict__ mask_q,
    bf16* __restrict__ s1, bf16* __restrict__ s2t) {
  __shared__ float mp[LEN_C];
  __shared__ float redA[4][64];
  __shared__ float redB[4][64];
  int b  = blockIdx.x >> 3;
  int j0 = (blockIdx.x & 7) * 64;
  int tid = threadIdx.x;
  for (int i = tid; i < LEN_C; i += 256) mp[i] = (float)mask_p[b * LEN_C + i];
  __syncthreads();
  int jl = tid & 63, sub = tid >> 6;
  int j = j0 + jl;
  const float* col = S + (size_t)b * LEN_C * LEN_Q + j;

  float m1 = -INFINITY, m2 = -INFINITY;
  for (int i = sub; i < LEN_C; i += 4) {
    float v = col[(size_t)i * LEN_Q];
    m1 = fmaxf(m1, v);
    m2 = (mp[i] != 0.0f) ? fmaxf(m2, v) : m2;
  }
  redA[sub][jl] = m1; redB[sub][jl] = m2;
  __syncthreads();
  m1 = fmaxf(fmaxf(redA[0][jl], redA[1][jl]), fmaxf(redA[2][jl], redA[3][jl]));
  m2 = fmaxf(fmaxf(redB[0][jl], redB[1][jl]), fmaxf(redB[2][jl], redB[3][jl]));
  __syncthreads();

  float sum1 = 0.f, sum2 = 0.f;
  for (int i = sub; i < LEN_C; i += 4) {
    float v = col[(size_t)i * LEN_Q];
    sum1 += __expf(v - m1);
    sum2 += (mp[i] != 0.0f) ? __expf(v - m2) : 0.f;
  }
  redA[sub][jl] = sum1; redB[sub][jl] = sum2;
  __syncthreads();
  sum1 = redA[0][jl] + redA[1][jl] + redA[2][jl] + redA[3][jl];
  sum2 = redB[0][jl] + redB[1][jl] + redB[2][jl] + redB[3][jl];

  const float unif = 1.0f / (float)LEN_C;
  bool qvalid = mask_q[b * LEN_Q + j] != 0;   // masked column -> uniform softmax
  bool pnone  = !(m2 > -INFINITY);            // whole column context-masked
  float inv1 = 1.0f / sum1;
  float inv2 = pnone ? 0.0f : 1.0f / sum2;
  bf16* s1c = s1  + (size_t)b * LEN_C * LEN_Q + j;
  bf16* s2r = s2t + ((size_t)b * LEN_Q + j) * LEN_C;
  for (int i = sub; i < LEN_C; i += 4) {
    float v = col[(size_t)i * LEN_Q];
    float o1 = qvalid ? __expf(v - m1) * inv1 : unif;
    float o2 = pnone ? unif : ((mp[i] != 0.0f) ? __expf(v - m2) * inv2 : 0.0f);
    s1c[(size_t)i * LEN_Q] = (bf16)o1;
    s2r[i] = (bf16)o2;
  }
}

// ---------------------------------------------------------------------------
// Kernel 4: a = s1 @ query.  A = s1 (L1 x L2), B from QT (D x L2). K = 512.
// ---------------------------------------------------------------------------
__global__ __launch_bounds__(256) void cqa_gemm_a(
    const bf16* __restrict__ s1, const bf16* __restrict__ QT,
    float* __restrict__ aOut) {
  __shared__ __align__(16) bf16 Btile[2 * BPANEL_E];
  int tid = threadIdx.x;
  int lane = tid & 31, wave = tid >> 5;
  int h = lane >> 4, r = lane & 15;
  int bi = blockIdx.x;
  int b  = bi >> 5;            // 16 mb * 2 nb = 32 blocks per batch
  int mb = (bi >> 1) & 15;
  int nb = bi & 1;
  int i0 = mb * 128 + wave * 16;
  int d0 = nb * 64;
  const bf16* A  = s1 + (size_t)b * LEN_C * LEN_Q;
  const bf16* Bm = QT + (size_t)b * DIM * LEN_Q;
  v8f acc[4] = {};
  gemm_lds_tile<LEN_Q / 32>(A, LEN_Q, i0, Bm, LEN_Q, d0, Btile, tid, acc);
  float* ab = aOut + (size_t)b * LEN_C * DIM;
#pragma unroll
  for (int nt = 0; nt < 4; ++nt)
#pragma unroll
    for (int rr = 0; rr < 8; ++rr)
      ab[(size_t)(i0 + rr + 8 * h) * DIM + d0 + nt * 16 + r] = acc[nt][rr];
}

// ---------------------------------------------------------------------------
// Kernel 5: t = s2^T @ ctx.  A = s2t (L2 x L1), B from CT (D x L1). K = 2048.
// Output stored transposed bf16: tT (D x L2) -> B operand of the b-GEMM.
// ---------------------------------------------------------------------------
__global__ __launch_bounds__(256) void cqa_gemm_t(
    const bf16* __restrict__ s2t, const bf16* __restrict__ CT,
    bf16* __restrict__ tT) {
  __shared__ __align__(16) bf16 Btile[2 * BPANEL_E];
  int tid = threadIdx.x;
  int lane = tid & 31, wave = tid >> 5;
  int h = lane >> 4, r = lane & 15;
  int bi = blockIdx.x;
  int b  = bi >> 3;            // 4 mb * 2 nb = 8 blocks per batch
  int mb = (bi >> 1) & 3;
  int nb = bi & 1;
  int j0 = mb * 128 + wave * 16;
  int d0 = nb * 64;
  const bf16* A  = s2t + (size_t)b * LEN_Q * LEN_C;
  const bf16* Bm = CT  + (size_t)b * DIM * LEN_C;
  v8f acc[4] = {};
  gemm_lds_tile<LEN_C / 32>(A, LEN_C, j0, Bm, LEN_C, d0, Btile, tid, acc);
#pragma unroll
  for (int nt = 0; nt < 4; ++nt) {
    int d = d0 + nt * 16 + r;
    v8bf o;
#pragma unroll
    for (int rr = 0; rr < 8; ++rr) o[rr] = (bf16)acc[nt][rr];
    *(v8bf*)(tT + ((size_t)b * DIM + d) * LEN_Q + j0 + 8 * h) = o;  // j contiguous
  }
}

// ---------------------------------------------------------------------------
// Kernel 6: bOut = s1 @ t, fused with the final concat epilogue:
//   out[b,i,:] = [ ctx | a | ctx*a | ctx*bOut ]   (4*D = 512 cols, f32)
// ---------------------------------------------------------------------------
__global__ __launch_bounds__(256) void cqa_gemm_b_out(
    const bf16* __restrict__ s1, const bf16* __restrict__ tT,
    const float* __restrict__ ctx, const float* __restrict__ aBuf,
    float* __restrict__ out) {
  __shared__ __align__(16) bf16 Btile[2 * BPANEL_E];
  int tid = threadIdx.x;
  int lane = tid & 31, wave = tid >> 5;
  int h = lane >> 4, r = lane & 15;
  int bi = blockIdx.x;
  int b  = bi >> 5;
  int mb = (bi >> 1) & 15;
  int nb = bi & 1;
  int i0 = mb * 128 + wave * 16;
  int d0 = nb * 64;
  const bf16* A  = s1 + (size_t)b * LEN_C * LEN_Q;
  const bf16* Bm = tT + (size_t)b * DIM * LEN_Q;
  v8f acc[4] = {};
  gemm_lds_tile<LEN_Q / 32>(A, LEN_Q, i0, Bm, LEN_Q, d0, Btile, tid, acc);
#pragma unroll
  for (int nt = 0; nt < 4; ++nt) {
#pragma unroll
    for (int rr = 0; rr < 8; ++rr) {
      int i = i0 + rr + 8 * h;
      int d = d0 + nt * 16 + r;
      size_t ibase = (size_t)b * LEN_C + i;
      float c  = ctx[ibase * DIM + d];
      float av = aBuf[ibase * DIM + d];
      float bv = acc[nt][rr];
      float* o = out + ibase * (4 * DIM);
      o[d]           = c;
      o[DIM + d]     = av;
      o[2 * DIM + d] = c * av;
      o[3 * DIM + d] = c * bv;
    }
  }
}

// ---------------------------------------------------------------------------
extern "C" void kernel_launch(void* const* d_in, const int* in_sizes, int n_in,
                              void* d_out, int out_size, void* d_ws, size_t ws_size,
                              hipStream_t stream) {
  const float* ctx    = (const float*)d_in[0];
  const float* qry    = (const float*)d_in[1];
  const float* w      = (const float*)d_in[2];
  const int*   mask_p = (const int*)d_in[3];
  const int*   mask_q = (const int*)d_in[4];
  float* out = (float*)d_out;

  char* ws = (char*)d_ws;
  size_t off = 0;
  auto carve = [&](size_t bytes) -> void* {
    void* p = ws + off;
    off += (bytes + 255) & ~(size_t)255;
    return p;
  };
  float* S    = (float*)carve((size_t)BATCH * LEN_C * LEN_Q * 4);  // 64 MiB
  bf16*  s1b  = (bf16*) carve((size_t)BATCH * LEN_C * LEN_Q * 2);  // 32 MiB
  bf16*  s2t  = (bf16*) carve((size_t)BATCH * LEN_Q * LEN_C * 2);  // 32 MiB
  bf16*  Abf  = (bf16*) carve((size_t)BATCH * LEN_C * DIM * 2);    //  8 MiB
  bf16*  Qbf  = (bf16*) carve((size_t)BATCH * LEN_Q * DIM * 2);    //  2 MiB
  bf16*  QT   = (bf16*) carve((size_t)BATCH * DIM * LEN_Q * 2);    //  2 MiB
  bf16*  CT   = (bf16*) carve((size_t)BATCH * DIM * LEN_C * 2);    //  8 MiB
  bf16*  tT   = (bf16*) carve((size_t)BATCH * DIM * LEN_Q * 2);    //  2 MiB
  float* aBuf = (float*)carve((size_t)BATCH * LEN_C * DIM * 4);    // 16 MiB
  float* cw   = (float*)carve((size_t)BATCH * LEN_C * 4);
  float* qw   = (float*)carve((size_t)BATCH * LEN_Q * 4);

  cqa_prep<<<5120, 256, 0, stream>>>(ctx, qry, w, Abf, Qbf, QT, CT, cw, qw);
  cqa_scores<<<2048, 256, 0, stream>>>(Abf, Qbf, cw, qw, S);
  cqa_softmax<<<BATCH * (LEN_Q / 64), 256, 0, stream>>>(S, mask_p, mask_q, s1b, s2t);
  cqa_gemm_a<<<512, 256, 0, stream>>>(s1b, QT, aBuf);
  cqa_gemm_t<<<128, 256, 0, stream>>>(s2t, CT, tT);
  cqa_gemm_b_out<<<512, 256, 0, stream>>>(s1b, tT, ctx, aBuf, out);

  (void)in_sizes; (void)n_in; (void)out_size; (void)ws_size;
}
